// RelativePositionalAttention_64716567216650
// MI455X (gfx1250) — matrix operand order repaired and checked by reading, not compiled, not run on hardware
//
#include <hip/hip_runtime.h>
#include <hip/hip_bf16.h>
#include <math.h>

// ---------------------------------------------------------------------------
// Problem constants (reference: B=2, S=2048, D=1024, N=16, H=64, T=2S)
// ---------------------------------------------------------------------------
#define BB   2
#define SS   2048
#define DD   1024
#define NN   16
#define HH   64
#define TT   4096
#define NORMF 0.125f            // 1/sqrt(64)
#define INFF  1000000.0f

typedef __bf16 bf16;
typedef __attribute__((ext_vector_type(16))) __bf16 v16bf;
typedef __attribute__((ext_vector_type(8)))  float  v8f;

union Frag { v16bf v; uint4 q[2]; };

__device__ __forceinline__ v8f wmma_bf16(v16bf a, v16bf b, v8f c) {
  // D = A(16x32 bf16) * B(32x16 bf16) + C(16x16 f32)
  return __builtin_amdgcn_wmma_f32_16x16x32_bf16(false, a, false, b,
                                                 (short)0, c, false, false);
}

__device__ __forceinline__ unsigned pack2(float a, float b) {
  union { __bf16 h[2]; unsigned u; } x;
  x.h[0] = (__bf16)a; x.h[1] = (__bf16)b;
  return x.u;
}

__device__ __forceinline__ v8f v8zero() {
  v8f z = {0.f,0.f,0.f,0.f,0.f,0.f,0.f,0.f};
  return z;
}

// Async global->LDS 16B copy (gfx1250 GLOBAL_LOAD_ASYNC_TO_LDS_B128, ASYNCcnt).
// LDS address passed as raw 32-bit offset VGPR (low 32 bits of generic pointer),
// global address as 64-bit VGPR pair.  Inline asm keeps it toolchain-portable.
__device__ __forceinline__ void async_copy_b128(const void* gptr, void* lptr) {
  const unsigned           lo = (unsigned)(uintptr_t)lptr;
  const unsigned long long ga = (unsigned long long)(uintptr_t)gptr;
  asm volatile("global_load_async_to_lds_b128 %0, %1, off"
               :: "v"(lo), "v"(ga) : "memory");
}
__device__ __forceinline__ void wait_asynccnt0() {
  asm volatile("s_wait_asynccnt 0x0" ::: "memory");
}

// ---------------------------------------------------------------------------
// Generic 4096 x 1024 x 1024 bf16-WMMA GEMM with mode-specific epilogues.
//   MODE 0: q projection  -> qh(f32, B,N,S,H), qw/qr bf16 (+bias*NORM)
//   MODE 1: k projection  -> kh bf16 (B,N,S,H), +kb
//   MODE 2: v projection  -> vh bf16 (B,N,H,S)  (transposed!), +vb
//   MODE 3: r projection  -> rh bf16 (N,T,H)
//   MODE 4: post GEMM     -> xbuf f32 = acc + postb + q   (residual fused)
// Block: 256 thr (8 waves, 4x2), tile 128x128, k-step 32.
// A tile staged f32 via async-DMA to LDS (converted during fragment build);
// B tile staged synchronously with transpose + f32->bf16 conversion.
// ---------------------------------------------------------------------------
template<int MODE>
__global__ __launch_bounds__(256)
void gemm_proj(const float* __restrict__ A, const float* __restrict__ Bw,
               float* __restrict__ out_f,
               bf16* __restrict__ oh0, bf16* __restrict__ oh1,
               const float* __restrict__ bias0, const float* __restrict__ bias1,
               const float* __restrict__ resid)
{
  __shared__ float lAf[128 * 36];  // [row][k] f32, stride 36 (144B, 16B-aligned)
  __shared__ bf16  lB [128 * 40];  // [col][k] bf16 (B transposed: K-contiguous)

  const int t    = threadIdx.x;
  const int m0   = blockIdx.y * 128;
  const int c0   = blockIdx.x * 128;
  const int wid  = t >> 5, lane = t & 31;
  const int wm   = wid >> 1;          // 0..3 : 32 rows each
  const int wn   = wid & 1;           // 0..1 : 64 cols each
  const int lrow = lane & 15;
  const int aklo = (lane & 16) ? 8 : 0;   // A fragment K-base within 32
  const int bklo = (lane & 16) ? 8 : 0;   // B fragment K-base within 32

  v8f acc[2][4];
  for (int i = 0; i < 2; ++i)
    for (int j = 0; j < 4; ++j) acc[i][j] = v8zero();

  for (int k0 = 0; k0 < DD; k0 += 32) {
    __syncthreads();   // previous-iteration LDS reads done before overwrite
    // ---- A tile 128x32 f32: async DMA straight into LDS (4 x b128 / thread)
    for (int s = t; s < 1024; s += 256) {
      const int row = s >> 3, kq = s & 7;
      async_copy_b128(A + (size_t)(m0 + row) * DD + k0 + kq * 4,
                      lAf + row * 36 + kq * 4);
    }
    // ---- B tile (transposed) 128 cols x 32 k, f32 -> bf16, overlaps A DMA
    {
      const int c = t & 127, kb0 = t >> 7;
      if (k0 + 32 < DD)
        __builtin_prefetch(Bw + (size_t)(k0 + 32 + kb0) * 1024 + c0 + c, 0, 1);
      for (int kk = 0; kk < 16; ++kk) {
        const int k = kb0 + kk * 2;
        lB[c * 40 + k] = (bf16)Bw[(size_t)(k0 + k) * 1024 + c0 + c];
      }
    }
    wait_asynccnt0();  // A DMA landed in LDS
    __syncthreads();
    // ---- A fragments: read f32 from LDS, pack to bf16
    Frag af[2];
    for (int tm = 0; tm < 2; ++tm) {
      const float* p = lAf + (wm * 32 + tm * 16 + lrow) * 36 + aklo;
      const float4 f0 = *(const float4*)p;
      const float4 f1 = *(const float4*)(p + 4);
      const float4 f2 = *(const float4*)(p + 16);
      const float4 f3 = *(const float4*)(p + 20);
      uint4 u0, u1;
      u0.x = pack2(f0.x, f0.y); u0.y = pack2(f0.z, f0.w);
      u0.z = pack2(f1.x, f1.y); u0.w = pack2(f1.z, f1.w);
      u1.x = pack2(f2.x, f2.y); u1.y = pack2(f2.z, f2.w);
      u1.z = pack2(f3.x, f3.y); u1.w = pack2(f3.z, f3.w);
      af[tm].q[0] = u0;
      af[tm].q[1] = u1;
    }
    // ---- B fragments (bf16, 2 x 16B LDS reads each)
    Frag bfr[4];
    for (int tn = 0; tn < 4; ++tn) {
      const bf16* p = lB + (wn * 64 + tn * 16 + lrow) * 40 + bklo;
      bfr[tn].q[0] = *(const uint4*)p;
      bfr[tn].q[1] = *(const uint4*)(p + 16);
    }
    for (int tm = 0; tm < 2; ++tm)
      for (int tn = 0; tn < 4; ++tn)
        acc[tm][tn] = wmma_bf16(af[tm].v, bfr[tn].v, acc[tm][tn]);
  }

  // ---- epilogue
  const int rhalf = (lane & 16) ? 8 : 0;
  for (int tm = 0; tm < 2; ++tm) {
    for (int tn = 0; tn < 4; ++tn) {
      const int colg = c0 + wn * 64 + tn * 16 + lrow;
      const int rb   = m0 + wm * 32 + tm * 16 + rhalf;
      for (int r = 0; r < 8; ++r) {
        const int   m = rb + r;
        const float v = acc[tm][tn][r];
        if constexpr (MODE == 0) {
          const int b = m >> 11, i = m & (SS - 1);
          const int n = colg >> 6, h = colg & 63;
          const size_t o = (((size_t)(b * NN + n)) * SS + i) * HH + h;
          const float qv = v * NORMF;
          out_f[o] = qv;
          oh0[o] = (bf16)(qv + NORMF * bias0[colg]);   // + r_w_bias*NORM
          oh1[o] = (bf16)(qv + NORMF * bias1[colg]);   // + r_r_bias*NORM
        } else if constexpr (MODE == 1) {
          const int b = m >> 11, i = m & (SS - 1);
          const int n = colg >> 6, h = colg & 63;
          oh0[(((size_t)(b * NN + n)) * SS + i) * HH + h] = (bf16)(v + bias0[colg]);
        } else if constexpr (MODE == 2) {
          const int b = m >> 11, i = m & (SS - 1);
          const int n = colg >> 6, h = colg & 63;
          oh0[(((size_t)(b * NN + n)) * HH + h) * SS + i] = (bf16)(v + bias0[colg]);
        } else if constexpr (MODE == 3) {
          const int n = colg >> 6, h = colg & 63;
          oh0[((size_t)n * TT + m) * HH + h] = (bf16)v;
        } else { // MODE 4
          out_f[(size_t)m * DD + colg] =
              v + bias0[colg] + resid[(size_t)m * DD + colg];
        }
      }
    }
  }
}

// ---------------------------------------------------------------------------
// Segment-bias precompute: sb[b,n,i,{diff,same}] = (qh + NORM*r_s_bias) . seg_embed[s]
// ---------------------------------------------------------------------------
__global__ __launch_bounds__(256)
void sb_kernel(const float* __restrict__ qh, const float* __restrict__ seg_embed,
               const float* __restrict__ rsb, float* __restrict__ sb)
{
  const int idx = blockIdx.x * 256 + threadIdx.x;   // (b*N+n)*S + i
  const int n   = (idx >> 11) & (NN - 1);
  const float* qr = qh + (size_t)idx * HH;
  const float* rb = rsb + n * HH;
  const float* s0 = seg_embed + n * HH;             // seg 0 (diff)
  const float* s1 = seg_embed + NN * HH + n * HH;   // seg 1 (same)
  float d0 = 0.f, d1 = 0.f;
  for (int h = 0; h < HH; ++h) {
    const float qv = qr[h] + NORMF * rb[h];
    d0 += qv * s0[h];
    d1 += qv * s1[h];
  }
  sb[(size_t)idx * 2 + 0] = d0;
  sb[(size_t)idx * 2 + 1] = d1;
}

// ---------------------------------------------------------------------------
// Fused flash attention with relative-shift positional bias + segment bias.
// Grid (S/64, N, B), block 128 (4 waves). Wave w owns i-tile i0 = bx*64+w*16
// and loops causal j-chunks of 32. rel_shift: pos[i,j] = bd[i, S+j-i].
// ---------------------------------------------------------------------------
__global__ __launch_bounds__(128)
void attn_kernel(const bf16* __restrict__ qw, const bf16* __restrict__ qr,
                 const bf16* __restrict__ kh, const bf16* __restrict__ vh,
                 const bf16* __restrict__ rh, const float* __restrict__ sb,
                 const unsigned char* __restrict__ segmat,
                 float* __restrict__ av)
{
  __shared__ bf16 ldsP[4 * 16 * 40];   // per-wave P staging (f32 C -> bf16 A frag)

  const int b    = blockIdx.z, n = blockIdx.y;
  const int wid  = threadIdx.x >> 5, lane = threadIdx.x & 31;
  const int i0   = blockIdx.x * 64 + wid * 16;
  const int lrow = lane & 15;
  const int hi   = (lane & 16) ? 1 : 0;     // high half owns rows +8
  const size_t headSH = ((size_t)(b * NN + n)) * SS * HH;  // (S,H) tensors
  const size_t headHS = ((size_t)(b * NN + n)) * HH * SS;  // (H,S) v tensor

  // A fragments: 16x64 split into two 16x32 h-halves. Row = i0+lrow,
  // K offsets {klo..klo+7, klo+16..klo+23}, klo = hi*8.
  Frag aw[2], ar[2];
  {
    const int irow = i0 + lrow, klo = hi * 8;
    for (int hh = 0; hh < 2; ++hh) {
      const bf16* pw = qw + headSH + (size_t)irow * HH + hh * 32 + klo;
      aw[hh].q[0] = *(const uint4*)pw;
      aw[hh].q[1] = *(const uint4*)(pw + 16);
      const bf16* pr = qr + headSH + (size_t)irow * HH + hh * 32 + klo;
      ar[hh].q[0] = *(const uint4*)pr;
      ar[hh].q[1] = *(const uint4*)(pr + 16);
    }
  }

  // Per-row segment bias (8 rows owned by this lane's half).
  float sbd[8], sbs[8];
  for (int r = 0; r < 8; ++r) {
    const float* p = sb + ((((size_t)(b * NN + n)) * SS) + i0 + hi * 8 + r) * 2;
    sbd[r] = p[0]; sbs[r] = p[1];
  }

  float m8[8], l8[8];
  v8f o[4];
  for (int r = 0; r < 8; ++r) { m8[r] = -1e30f; l8[r] = 0.f; }
  for (int ht = 0; ht < 4; ++ht) o[ht] = v8zero();

  const int bsel = (lane & 16) ? 8 : 0;    // B fragment K base

  for (int j0 = 0; j0 <= i0 + 15; j0 += 32) {
    // ---- content scores: two 16x16 tiles over j0..j0+31
    v8f sc[2] = { v8zero(), v8zero() };
    for (int jt = 0; jt < 2; ++jt) {
      const int j = j0 + jt * 16 + lrow;
      for (int hh = 0; hh < 2; ++hh) {
        Frag kb_;
        const bf16* p = kh + headSH + (size_t)j * HH + hh * 32 + bsel;
        kb_.q[0] = *(const uint4*)p;
        kb_.q[1] = *(const uint4*)(p + 16);
        sc[jt] = wmma_bf16(aw[hh].v, kb_.v, sc[jt]);
      }
    }
    // ---- bd tiles: t-window [tmin, tmin+47], 3 tiles of 16 columns
    v8f bd[3] = { v8zero(), v8zero(), v8zero() };
    const int tmin = SS + j0 - i0 - 15;
    for (int ct = 0; ct < 3; ++ct) {
      const int tcol = tmin + ct * 16 + lrow;
      for (int hh = 0; hh < 2; ++hh) {
        Frag rb_;
        const bf16* p = rh + ((size_t)n * TT + tcol) * HH + hh * 32 + bsel;
        rb_.q[0] = *(const uint4*)p;
        rb_.q[1] = *(const uint4*)(p + 16);
        bd[ct] = wmma_bf16(ar[hh].v, rb_.v, bd[ct]);
      }
    }
    // ---- combine + online softmax (row r lives in a 16-lane half)
    float ps_[2][8];
    for (int r = 0; r < 8; ++r) {
      const int ig = i0 + hi * 8 + r;
      float mloc = -1e30f;
      for (int jt = 0; jt < 2; ++jt) {
        const int jl = jt * 16 + lrow;
        const int jg = j0 + jl;
        // rel-shift diagonal gather: column in bd-window, same VGPR, lane permute
        const int c    = jl + 15 - (hi * 8 + r);      // 0..46
        const int tile = c >> 4;
        const int src  = (c & 15) + (lane & 16);
        const float p0 = __shfl(bd[0][r], src, 32);
        const float p1 = __shfl(bd[1][r], src, 32);
        const float p2 = __shfl(bd[2][r], src, 32);
        const float pos = (tile == 0) ? p0 : ((tile == 1) ? p1 : p2);
        const unsigned char sm = segmat[((size_t)b * SS + ig) * SS + jg];
        float s = sc[jt][r] + pos + (sm ? sbs[r] : sbd[r]);
        if (jg > ig) s -= INFF;                        // causal mask (== attn_mask)
        ps_[jt][r] = s;
        mloc = fmaxf(mloc, s);
      }
      for (int off = 1; off < 16; off <<= 1)
        mloc = fmaxf(mloc, __shfl_xor(mloc, off, 32));
      const float mnew  = fmaxf(m8[r], mloc);
      const float scale = __expf(m8[r] - mnew);
      float psum = 0.f;
      for (int jt = 0; jt < 2; ++jt) {
        const float e = __expf(ps_[jt][r] - mnew);
        ps_[jt][r] = e; psum += e;
      }
      for (int off = 1; off < 16; off <<= 1)
        psum += __shfl_xor(psum, off, 32);
      l8[r] = l8[r] * scale + psum;
      m8[r] = mnew;
      for (int ht = 0; ht < 4; ++ht) o[ht][r] *= scale;
    }
    // ---- P (C-layout f32) -> LDS bf16 -> A fragment (wave-private, DS in-order)
    bf16* lp = ldsP + wid * 16 * 40;
    for (int jt = 0; jt < 2; ++jt)
      for (int r = 0; r < 8; ++r)
        lp[(hi * 8 + r) * 40 + jt * 16 + lrow] = (bf16)ps_[jt][r];
    Frag pa;
    {
      const bf16* p = lp + lrow * 40 + hi * 8;
      pa.q[0] = *(const uint4*)p;
      pa.q[1] = *(const uint4*)(p + 16);
    }
    // ---- O += P(16x32) * V(32x64), V stored (H,S) so B frags are contiguous
    for (int ht = 0; ht < 4; ++ht) {
      const int h = ht * 16 + lrow;
      Frag vb_;
      const bf16* p = vh + headHS + (size_t)h * SS + j0 + bsel;
      vb_.q[0] = *(const uint4*)p;
      vb_.q[1] = *(const uint4*)(p + 16);
      o[ht] = wmma_bf16(pa.v, vb_.v, o[ht]);
    }
  }

  // ---- normalize + write attn_vec (B,S,N*H)
  for (int ht = 0; ht < 4; ++ht) {
    const int h = n * HH + ht * 16 + lrow;
    for (int r = 0; r < 8; ++r) {
      const int ig = i0 + hi * 8 + r;
      av[((size_t)b * SS + ig) * DD + h] = o[ht][r] / l8[r];
    }
  }
}

// ---------------------------------------------------------------------------
// LayerNorm over D=1024 (input xbuf = q + attn_out, residual already fused).
// ---------------------------------------------------------------------------
__global__ __launch_bounds__(256)
void ln_kernel(const float* __restrict__ x, const float* __restrict__ g,
               const float* __restrict__ bta, float* __restrict__ out)
{
  __shared__ float red[2][8];
  const int row = blockIdx.x;
  const float* xr = x + (size_t)row * DD;
  float s = 0.f, s2 = 0.f;
  for (int c = threadIdx.x; c < DD; c += 256) {
    const float v = xr[c];
    s += v; s2 += v * v;
  }
  for (int off = 16; off; off >>= 1) {
    s  += __shfl_xor(s,  off, 32);
    s2 += __shfl_xor(s2, off, 32);
  }
  const int wid = threadIdx.x >> 5, lane = threadIdx.x & 31;
  if (lane == 0) { red[0][wid] = s; red[1][wid] = s2; }
  __syncthreads();
  s = 0.f; s2 = 0.f;
  for (int w = 0; w < 8; ++w) { s += red[0][w]; s2 += red[1][w]; }
  const float mu  = s * (1.0f / DD);
  const float var = s2 * (1.0f / DD) - mu * mu;
  const float rs  = rsqrtf(var + 1e-5f);
  for (int c = threadIdx.x; c < DD; c += 256)
    out[(size_t)row * DD + c] = (xr[c] - mu) * rs * g[c] + bta[c];
}

// ---------------------------------------------------------------------------
// Host side
// ---------------------------------------------------------------------------
extern "C" void kernel_launch(void* const* d_in, const int* in_sizes, int n_in,
                              void* d_out, int out_size, void* d_ws, size_t ws_size,
                              hipStream_t stream)
{
  (void)in_sizes; (void)n_in; (void)out_size; (void)ws_size;
  const float* q        = (const float*)d_in[0];
  const float* k        = (const float*)d_in[1];
  const float* v        = (const float*)d_in[2];
  const float* pos_enc  = (const float*)d_in[3];
  const unsigned char* seg_mat = (const unsigned char*)d_in[4];  // bool (B,S,S)
  /* d_in[5] = attn_mask: exactly the causal mask; applied arithmetically */
  const float* qW       = (const float*)d_in[6];
  const float* kW       = (const float*)d_in[7];
  const float* kb       = (const float*)d_in[8];
  const float* vW       = (const float*)d_in[9];
  const float* vb       = (const float*)d_in[10];
  const float* r_w_bias = (const float*)d_in[11];
  const float* r_r_bias = (const float*)d_in[12];
  const float* r_kernel = (const float*)d_in[13];
  const float* r_s_bias = (const float*)d_in[14];
  const float* seg_embed= (const float*)d_in[15];
  const float* postW    = (const float*)d_in[16];
  const float* postb    = (const float*)d_in[17];
  const float* ln_g     = (const float*)d_in[18];
  const float* ln_b     = (const float*)d_in[19];

  // workspace carve-out (~93 MB total)
  char* ws = (char*)d_ws;
  size_t off = 0;
  auto carve = [&](size_t bytes) {
    char* p = ws + off;
    off = (off + bytes + 255) & ~(size_t)255;
    return p;
  };
  const size_t EH = (size_t)BB * NN * SS * HH;   // 4,194,304 head elements
  float* qh   = (float*)carve(EH * 4);           // (B,N,S,H) f32
  bf16*  qwb  = (bf16*) carve(EH * 2);           // (B,N,S,H) +r_w_bias*NORM
  bf16*  qrb  = (bf16*) carve(EH * 2);           // (B,N,S,H) +r_r_bias*NORM
  bf16*  khb  = (bf16*) carve(EH * 2);           // (B,N,S,H)
  bf16*  vhb  = (bf16*) carve(EH * 2);           // (B,N,H,S) transposed
  bf16*  rhb  = (bf16*) carve((size_t)NN * TT * HH * 2); // (N,T,H)
  float* sbuf = (float*)carve((size_t)BB * NN * SS * 2 * 4);
  float* av   = (float*)carve((size_t)BB * SS * DD * 4);
  float* xbuf = (float*)carve((size_t)BB * SS * DD * 4);

  const dim3 gG(8, 32);  // 128-col x 128-row tiles over 4096x1024 output
  gemm_proj<0><<<gG, 256, 0, stream>>>(q, qW, qh, qwb, qrb, r_w_bias, r_r_bias, nullptr);
  gemm_proj<1><<<gG, 256, 0, stream>>>(k, kW, nullptr, khb, nullptr, kb, nullptr, nullptr);
  gemm_proj<2><<<gG, 256, 0, stream>>>(v, vW, nullptr, vhb, nullptr, vb, nullptr, nullptr);
  gemm_proj<3><<<gG, 256, 0, stream>>>(pos_enc, r_kernel, nullptr, rhb, nullptr, nullptr, nullptr, nullptr);

  sb_kernel<<<(BB * NN * SS) / 256, 256, 0, stream>>>(qh, seg_embed, r_s_bias, sbuf);

  attn_kernel<<<dim3(SS / 64, NN, BB), 128, 0, stream>>>(qwb, qrb, khb, vhb, rhb,
                                                         sbuf, seg_mat, av);

  gemm_proj<4><<<gG, 256, 0, stream>>>(av, postW, xbuf, nullptr, nullptr, postb, nullptr, q);

  ln_kernel<<<BB * SS, 256, 0, stream>>>(xbuf, ln_g, ln_b, (float*)d_out);
}
